// TransformerBlock_283467842664
// MI455X (gfx1250) — compile-verified, matrix operand order checked by reading, not run on hardware
//
#include <hip/hip_runtime.h>
#include <cstddef>

// ---------------------------------------------------------------------------
// Types for CDNA5 WMMA (wave32, v_wmma_f32_16x16x32_bf16)
// ---------------------------------------------------------------------------
typedef __attribute__((ext_vector_type(16))) __bf16     v16bf;
typedef __attribute__((ext_vector_type(8)))  float      v8f;
typedef __attribute__((ext_vector_type(8)))  unsigned short u16x8;

union BFrag {
    v16bf          v;
    u16x8          h[2];
    unsigned short u[16];
};

__device__ inline unsigned short f32_to_bf16(float f) {
    union { float f; unsigned int u; } a;
    a.f = f;
    unsigned int u = a.u;
    u += 0x7FFFu + ((u >> 16) & 1u);     // round-to-nearest-even
    return (unsigned short)(u >> 16);
}

__device__ inline v8f wmma_bf16(const BFrag& a, const BFrag& b, v8f c) {
    return __builtin_amdgcn_wmma_f32_16x16x32_bf16(
        /*neg_a=*/false, a.v, /*neg_b=*/false, b.v,
        /*c_mod=*/(short)0, c, /*reuse_a=*/false, /*reuse_b=*/false);
}

__device__ inline int iclamp(int v, int lo, int hi) {
    return v < lo ? lo : (v > hi ? hi : v);
}

// ---------------------------------------------------------------------------
// Problem constants
// ---------------------------------------------------------------------------
#define BATCH    64
#define SEQ      197
#define TOKENS   (BATCH * SEQ)      // 12608 = 197 * 64
#define EMBED    768
#define NHEADS   12
#define HDIM     64
#define QKV_N    (3 * EMBED)        // 2304
#define MLP_HID  (4 * EMBED)        // 3072
#define LA_HEADS 16
#define LA_G     (EMBED / LA_HEADS) // 48

// ---------------------------------------------------------------------------
// Workspace layout (bytes). Total = 150,519,808 B (~143.6 MB).
// h-buffer aliases xn; fc1 output (gelu, 12608x3072 bf16 = 77,463,552 B)
// exactly aliases the qkv+o span (58,097,664 + 19,365,888).
// ---------------------------------------------------------------------------
static constexpr size_t SZ_WQKV  = (size_t)QKV_N   * EMBED   * 2;
static constexpr size_t SZ_WPROJ = (size_t)EMBED   * EMBED   * 2;
static constexpr size_t SZ_WFC1  = (size_t)MLP_HID * EMBED   * 2;
static constexpr size_t SZ_WFC2  = (size_t)EMBED   * MLP_HID * 2;
static constexpr size_t SZ_GATE  = (size_t)BATCH * LA_HEADS * 196 * 4;
static constexpr size_t SZ_XA    = (size_t)TOKENS * EMBED * 4;
static constexpr size_t SZ_XN    = (size_t)TOKENS * EMBED * 2;
static constexpr size_t SZ_QKV   = (size_t)TOKENS * QKV_N * 2;

static constexpr size_t OFF_WQKV  = 0;
static constexpr size_t OFF_WPROJ = OFF_WQKV  + SZ_WQKV;
static constexpr size_t OFF_WFC1  = OFF_WPROJ + SZ_WPROJ;
static constexpr size_t OFF_WFC2  = OFF_WFC1  + SZ_WFC1;
static constexpr size_t OFF_GATE  = OFF_WFC2  + SZ_WFC2;
static constexpr size_t OFF_XA    = OFF_GATE  + SZ_GATE;
static constexpr size_t OFF_XN    = OFF_XA    + SZ_XA;     // also h (LN2 out)
static constexpr size_t OFF_QKV   = OFF_XN    + SZ_XN;     // also g (fc1 out)
static constexpr size_t OFF_O     = OFF_QKV   + SZ_QKV;

// ---------------------------------------------------------------------------
// fp32 -> bf16 weight conversion
// ---------------------------------------------------------------------------
__global__ void cvt_bf16_kernel(const float* __restrict__ src,
                                unsigned short* __restrict__ dst, int n) {
    int i = blockIdx.x * blockDim.x + threadIdx.x;
    if (i < n) dst[i] = f32_to_bf16(src[i]);
}

// ---------------------------------------------------------------------------
// LA gate: pooled stats -> channel MLP -> bilinear 7->14 -> sigmoid
// gate[b][h][196], one block per image
// ---------------------------------------------------------------------------
__global__ __launch_bounds__(256) void la_gate_kernel(
    const float* __restrict__ x,     // [64][197][768]
    const float* __restrict__ f1_w, const float* __restrict__ f1_b,
    const float* __restrict__ f2_w, const float* __restrict__ f2_b,
    const float* __restrict__ v_w,  const float* __restrict__ v_b,
    float* __restrict__ gate)        // [64][16][196]
{
    const int b = blockIdx.x;
    __shared__ float sm[LA_HEADS][49];
    __shared__ float sx[LA_HEADS][49];
    __shared__ float fus[LA_HEADS][49];
    __shared__ float hm[49], hx[49];

    const float* xb = x + ((size_t)b * SEQ + 1) * EMBED;   // spatial tokens

    for (int cell = threadIdx.x; cell < LA_HEADS * 49; cell += 256) {
        int h = cell / 49, ij = cell % 49;
        int ci = ij / 7, cj = ij % 7;
        float sum = 0.0f, mx = -3.4e38f;
        for (int gc = 0; gc < LA_G; ++gc) {
            int c = h * LA_G + gc;
            #pragma unroll
            for (int di = 0; di < 2; ++di)
                #pragma unroll
                for (int dj = 0; dj < 2; ++dj) {
                    int s = (2 * ci + di) * 14 + (2 * cj + dj);
                    float v = xb[(size_t)s * EMBED + c];
                    sum += v;
                    mx = fmaxf(mx, v);
                }
        }
        sm[h][ij] = sum * (1.0f / 192.0f);
        sx[h][ij] = mx;
    }
    __syncthreads();

    if (threadIdx.x < 49) {
        float am = 0.0f, ax = 0.0f;
        #pragma unroll
        for (int h = 0; h < LA_HEADS; ++h) {
            am += sm[h][threadIdx.x] * f1_w[h];
            ax += sx[h][threadIdx.x] * f1_w[h];
        }
        hm[threadIdx.x] = fmaxf(am + f1_b[0], 0.0f);
        hx[threadIdx.x] = fmaxf(ax + f1_b[0], 0.0f);
    }
    __syncthreads();

    for (int cell = threadIdx.x; cell < LA_HEADS * 49; cell += 256) {
        int h = cell / 49, ij = cell % 49;
        float m  = hm[ij] * f2_w[h] + f2_b[h];
        float mx = hx[ij] * f2_w[h] + f2_b[h];
        fus[h][ij] = v_w[0] * m + v_w[1] * mx + v_b[0];
    }
    __syncthreads();

    // half-pixel bilinear 7x7 -> 14x14 with edge clamp, then sigmoid
    for (int t = threadIdx.x; t < LA_HEADS * 196; t += 256) {
        int h = t / 196, p = t % 196;
        int oi = p / 14, oj = p % 14;
        float ci = 0.5f * oi - 0.25f;
        float cj = 0.5f * oj - 0.25f;
        int i0 = (int)floorf(ci); float wi = ci - (float)i0;
        int j0 = (int)floorf(cj); float wj = cj - (float)j0;
        int i0c = iclamp(i0, 0, 6), i1c = iclamp(i0 + 1, 0, 6);
        int j0c = iclamp(j0, 0, 6), j1c = iclamp(j0 + 1, 0, 6);
        float f00 = fus[h][i0c * 7 + j0c], f01 = fus[h][i0c * 7 + j1c];
        float f10 = fus[h][i1c * 7 + j0c], f11 = fus[h][i1c * 7 + j1c];
        float fv = (1.0f - wi) * ((1.0f - wj) * f00 + wj * f01)
                 +         wi  * ((1.0f - wj) * f10 + wj * f11);
        gate[((size_t)b * LA_HEADS + h) * 196 + p] = 1.0f / (1.0f + expf(-fv));
    }
}

// ---------------------------------------------------------------------------
// LayerNorm (optionally fused with LA gate apply: v = x*(1+gate)), bf16 out
// one block per token row
// ---------------------------------------------------------------------------
__global__ __launch_bounds__(256) void ln_kernel(
    const float* __restrict__ in,      // [12608][768]
    const float* __restrict__ gate,    // [64][16][196] or nullptr
    const float* __restrict__ gamma, const float* __restrict__ beta,
    unsigned short* __restrict__ out)  // [12608][768] bf16
{
    const int row = blockIdx.x;
    const int b = row / SEQ, n = row % SEQ;
    __shared__ float red[256];

    const float* xr = in + (size_t)row * EMBED;
    float vals[3];
    #pragma unroll
    for (int i = 0; i < 3; ++i) {
        int c = threadIdx.x + i * 256;
        float v = xr[c];
        if (gate != nullptr && n > 0) {
            int head = c / LA_G;
            float g = gate[((size_t)b * LA_HEADS + head) * 196 + (n - 1)];
            v = v * (1.0f + g);          // x*gate + x
        }
        vals[i] = v;
    }

    red[threadIdx.x] = vals[0] + vals[1] + vals[2];
    __syncthreads();
    for (int off = 128; off > 0; off >>= 1) {
        if (threadIdx.x < off) red[threadIdx.x] += red[threadIdx.x + off];
        __syncthreads();
    }
    float mu = red[0] * (1.0f / EMBED);
    __syncthreads();

    float d0 = vals[0] - mu, d1 = vals[1] - mu, d2 = vals[2] - mu;
    red[threadIdx.x] = d0 * d0 + d1 * d1 + d2 * d2;
    __syncthreads();
    for (int off = 128; off > 0; off >>= 1) {
        if (threadIdx.x < off) red[threadIdx.x] += red[threadIdx.x + off];
        __syncthreads();
    }
    float rstd = rsqrtf(red[0] * (1.0f / EMBED) + 1e-5f);

    #pragma unroll
    for (int i = 0; i < 3; ++i) {
        int c = threadIdx.x + i * 256;
        float y = (vals[i] - mu) * rstd * gamma[c] + beta[c];
        out[(size_t)row * EMBED + c] = f32_to_bf16(y);
    }
}

// ---------------------------------------------------------------------------
// Generic bf16 WMMA GEMM:  C[M,N] = A[M,K] @ W[N,K]^T (+bias, epilogue)
// A row-major bf16, W row-major (out,in) bf16 -> W rows are B-matrix columns,
// so A and B fragments load with the identical along-K pattern.
// block = 128 threads (4 waves), block tile 64(M) x 256(N),
// wave tile 64x64 = 4x4 fragments of 16x16 (16 accums), K stepped by 32.
// -> 16 WMMAs per 16 b128 loads per k-step (~32 FLOP/B from L2).
// mode: 0 = bf16 out (+bias)         [qkv]
//       1 = f32 out (+bias+residual) [proj, fc2]
//       2 = bf16 out gelu(+bias)     [fc1]
// Requires: M % 64 == 0, N % 256 == 0, K % 32 == 0 (true for all calls).
// ---------------------------------------------------------------------------
__global__ __launch_bounds__(128) void gemm_bf16_kernel(
    const unsigned short* __restrict__ A,
    const unsigned short* __restrict__ W,
    const float* __restrict__ bias,
    const float* __restrict__ res,
    void* __restrict__ out,
    int M, int N, int K, int mode)
{
    const int lane  = threadIdx.x & 31;
    const int wave  = threadIdx.x >> 5;             // 0..3
    const int rowBase = blockIdx.x * 64;
    const int colBase = blockIdx.y * 256 + wave * 64;
    const int lr    = lane & 15;
    const int khalf = (lane < 16) ? 0 : 8;

    const v8f zacc = {0.f, 0.f, 0.f, 0.f, 0.f, 0.f, 0.f, 0.f};
    v8f acc[4][4];
    #pragma unroll
    for (int mi = 0; mi < 4; ++mi)
        #pragma unroll
        for (int ni = 0; ni < 4; ++ni) acc[mi][ni] = zacc;

    const unsigned short* Abase = A + (size_t)(rowBase + lr) * K + khalf;
    const unsigned short* Wbase = W + (size_t)(colBase + lr) * K + khalf;

    for (int k0 = 0; k0 < K; k0 += 32) {
        BFrag a[4], bfr[4];
        #pragma unroll
        for (int mi = 0; mi < 4; ++mi) {
            const unsigned short* p = Abase + (size_t)(mi * 16) * K + k0;
            a[mi].h[0] = *(const u16x8*)(p);
            a[mi].h[1] = *(const u16x8*)(p + 16);
        }
        #pragma unroll
        for (int ni = 0; ni < 4; ++ni) {
            const unsigned short* p = Wbase + (size_t)(ni * 16) * K + k0;
            bfr[ni].h[0] = *(const u16x8*)(p);
            bfr[ni].h[1] = *(const u16x8*)(p + 16);
        }
        #pragma unroll
        for (int mi = 0; mi < 4; ++mi)
            #pragma unroll
            for (int ni = 0; ni < 4; ++ni)
                acc[mi][ni] = wmma_bf16(a[mi], bfr[ni], acc[mi][ni]);
    }

    // epilogue: C/D layout: lane col = lane%16; VGPR r -> row r + 8*(lane/16)
    const int rhi = (lane >> 4) << 3;
    #pragma unroll
    for (int mi = 0; mi < 4; ++mi) {
        #pragma unroll
        for (int ni = 0; ni < 4; ++ni) {
            int col = colBase + ni * 16 + lr;
            float bi = bias[col];
            #pragma unroll
            for (int r = 0; r < 8; ++r) {
                int row = rowBase + mi * 16 + r + rhi;
                size_t idx = (size_t)row * N + col;
                float v = acc[mi][ni][r] + bi;
                if (mode == 0) {
                    ((unsigned short*)out)[idx] = f32_to_bf16(v);
                } else if (mode == 1) {
                    ((float*)out)[idx] = v + res[idx];
                } else {
                    float g = 0.5f * v * (1.0f + erff(v * 0.70710678f));
                    ((unsigned short*)out)[idx] = f32_to_bf16(g);
                }
            }
        }
    }
}

// ---------------------------------------------------------------------------
// Attention: one wave per (b, head, 16-query tile). N padded 197 -> 224.
// S = Q K^T (WMMA, K=64 split in two 32-steps), softmax in LDS,
// O = P V (WMMA over 7 tiles of 32 score-columns).
// ---------------------------------------------------------------------------
#define QTILES 13           // ceil(197/16)
#define NPAD   224          // 14*16 = 7*32

__global__ __launch_bounds__(32) void attn_kernel(
    const unsigned short* __restrict__ qkv,  // [12608][2304] bf16
    unsigned short* __restrict__ o)          // [12608][768]  bf16
{
    const int blk  = blockIdx.x;
    const int qt   = blk % QTILES;
    const int h    = (blk / QTILES) % NHEADS;
    const int b    = blk / (QTILES * NHEADS);
    const int lane = threadIdx.x;
    const int lr   = lane & 15;
    const int khalf = (lane < 16) ? 0 : 8;
    const int rhi  = (lane >> 4) << 3;
    const float scale = 0.125f;              // 1/sqrt(64)

    __shared__ float          S[16][NPAD];
    __shared__ unsigned short P[16][NPAD];

    const u16x8 z8 = {0, 0, 0, 0, 0, 0, 0, 0};

    // Q fragments (reused for all key tiles)
    BFrag qa[2];
    {
        int qtok = qt * 16 + lr;
        bool qok = qtok < SEQ;
        const unsigned short* qrow =
            qkv + (size_t)(b * SEQ + (qok ? qtok : 0)) * QKV_N + h * HDIM;
        #pragma unroll
        for (int kk = 0; kk < 2; ++kk) {
            if (qok) {
                const unsigned short* p = qrow + kk * 32 + khalf;
                qa[kk].h[0] = *(const u16x8*)(p);
                qa[kk].h[1] = *(const u16x8*)(p + 16);
            } else {
                qa[kk].h[0] = z8; qa[kk].h[1] = z8;
            }
        }
    }

    // Phase 1: scores into LDS
    for (int nt = 0; nt < NPAD / 16; ++nt) {
        v8f acc = {};
        BFrag kb[2];
        int ktok = nt * 16 + lr;
        bool kok = ktok < SEQ;
        const unsigned short* krow =
            qkv + (size_t)(b * SEQ + (kok ? ktok : 0)) * QKV_N + EMBED + h * HDIM;
        #pragma unroll
        for (int kk = 0; kk < 2; ++kk) {
            if (kok) {
                const unsigned short* p = krow + kk * 32 + khalf;
                kb[kk].h[0] = *(const u16x8*)(p);
                kb[kk].h[1] = *(const u16x8*)(p + 16);
            } else {
                kb[kk].h[0] = z8; kb[kk].h[1] = z8;
            }
        }
        acc = wmma_bf16(qa[0], kb[0], acc);
        acc = wmma_bf16(qa[1], kb[1], acc);

        int col = nt * 16 + lr;
        #pragma unroll
        for (int r = 0; r < 8; ++r) {
            int m = r + rhi;
            S[m][col] = (col < SEQ) ? acc[r] * scale : -1e30f;
        }
    }
    __syncthreads();

    // Phase 2: row softmax (lanes 0..15 each own a query row)
    if (lane < 16) {
        int m = lane;
        float mx = -3.4e38f;
        for (int j = 0; j < NPAD; ++j) mx = fmaxf(mx, S[m][j]);
        float sum = 0.0f;
        for (int j = 0; j < NPAD; ++j) {
            float e = expf(S[m][j] - mx);
            S[m][j] = e;
            sum += e;
        }
        float ri = 1.0f / sum;
        for (int j = 0; j < NPAD; ++j)
            P[m][j] = f32_to_bf16(S[m][j] * ri);
    }
    __syncthreads();

    // Phase 3: O = P V
    v8f oacc[4];
    {
        const v8f z = {0.f, 0.f, 0.f, 0.f, 0.f, 0.f, 0.f, 0.f};
        #pragma unroll
        for (int dt = 0; dt < 4; ++dt) oacc[dt] = z;
    }

    for (int ct = 0; ct < NPAD / 32; ++ct) {
        BFrag pa;
        #pragma unroll
        for (int e = 0; e < 16; ++e) {
            int koff = khalf + (e < 8 ? e : e + 8);   // {khalf..+7, khalf+16..+23}
            pa.u[e] = P[lr][ct * 32 + koff];
        }
        #pragma unroll
        for (int dt = 0; dt < 4; ++dt) {
            BFrag vb;
            int dcol = dt * 16 + lr;
            #pragma unroll
            for (int e = 0; e < 16; ++e) {
                int koff = khalf + (e < 8 ? e : e + 8);
                int vtok = ct * 32 + koff;
                vb.u[e] = (vtok < SEQ)
                    ? qkv[(size_t)(b * SEQ + vtok) * QKV_N + 2 * EMBED + h * HDIM + dcol]
                    : (unsigned short)0;
            }
            oacc[dt] = wmma_bf16(pa, vb, oacc[dt]);
        }
    }

    // store O (heads concatenated back at h*64+d)
    #pragma unroll
    for (int dt = 0; dt < 4; ++dt) {
        int dcol = h * HDIM + dt * 16 + lr;
        #pragma unroll
        for (int r = 0; r < 8; ++r) {
            int m = r + rhi;
            int tok = qt * 16 + m;
            if (tok < SEQ)
                o[(size_t)(b * SEQ + tok) * EMBED + dcol] = f32_to_bf16(oacc[dt][r]);
        }
    }
}

// ---------------------------------------------------------------------------
// Launch
// ---------------------------------------------------------------------------
extern "C" void kernel_launch(void* const* d_in, const int* in_sizes, int n_in,
                              void* d_out, int out_size, void* d_ws, size_t ws_size,
                              hipStream_t stream) {
    (void)in_sizes; (void)n_in; (void)out_size; (void)ws_size;

    const float* x      = (const float*)d_in[0];
    const float* n1_g   = (const float*)d_in[1];
    const float* n1_b   = (const float*)d_in[2];
    const float* qkv_w  = (const float*)d_in[3];
    const float* qkv_b  = (const float*)d_in[4];
    const float* proj_w = (const float*)d_in[5];
    const float* proj_b = (const float*)d_in[6];
    const float* n2_g   = (const float*)d_in[7];
    const float* n2_b   = (const float*)d_in[8];
    const float* fc1_w  = (const float*)d_in[9];
    const float* fc1_b  = (const float*)d_in[10];
    const float* fc2_w  = (const float*)d_in[11];
    const float* fc2_b  = (const float*)d_in[12];
    const float* f1_w   = (const float*)d_in[13];
    const float* f1_b   = (const float*)d_in[14];
    const float* f2_w   = (const float*)d_in[15];
    const float* f2_b   = (const float*)d_in[16];
    const float* v_w    = (const float*)d_in[17];
    const float* v_b    = (const float*)d_in[18];

    char* ws = (char*)d_ws;
    unsigned short* wqkv  = (unsigned short*)(ws + OFF_WQKV);
    unsigned short* wproj = (unsigned short*)(ws + OFF_WPROJ);
    unsigned short* wfc1  = (unsigned short*)(ws + OFF_WFC1);
    unsigned short* wfc2  = (unsigned short*)(ws + OFF_WFC2);
    float*          gate  = (float*)         (ws + OFF_GATE);
    float*          xa    = (float*)         (ws + OFF_XA);
    unsigned short* xn    = (unsigned short*)(ws + OFF_XN);   // ln1 out
    unsigned short* hbuf  = (unsigned short*)(ws + OFF_XN);   // ln2 out (alias)
    unsigned short* qkvb  = (unsigned short*)(ws + OFF_QKV);
    unsigned short* gbuf  = (unsigned short*)(ws + OFF_QKV);  // fc1 out (alias qkv+o)
    unsigned short* obuf  = (unsigned short*)(ws + OFF_O);

    float* out = (float*)d_out;

    // 1. weights -> bf16
    {
        int n;
        n = QKV_N * EMBED;
        cvt_bf16_kernel<<<(n + 255) / 256, 256, 0, stream>>>(qkv_w, wqkv, n);
        n = EMBED * EMBED;
        cvt_bf16_kernel<<<(n + 255) / 256, 256, 0, stream>>>(proj_w, wproj, n);
        n = MLP_HID * EMBED;
        cvt_bf16_kernel<<<(n + 255) / 256, 256, 0, stream>>>(fc1_w, wfc1, n);
        n = EMBED * MLP_HID;
        cvt_bf16_kernel<<<(n + 255) / 256, 256, 0, stream>>>(fc2_w, wfc2, n);
    }

    // 2. LA gate
    la_gate_kernel<<<BATCH, 256, 0, stream>>>(x, f1_w, f1_b, f2_w, f2_b, v_w, v_b, gate);

    // 3. gate-apply + LN1 -> xn (bf16)
    ln_kernel<<<TOKENS, 256, 0, stream>>>(x, gate, n1_g, n1_b, xn);

    // 4. QKV = xn @ Wqkv^T + b  -> bf16
    {
        dim3 g(TOKENS / 64, QKV_N / 256);
        gemm_bf16_kernel<<<g, 128, 0, stream>>>(xn, wqkv, qkv_b, nullptr,
                                                qkvb, TOKENS, QKV_N, EMBED, 0);
    }

    // 5. attention -> obuf (bf16)
    attn_kernel<<<BATCH * NHEADS * QTILES, 32, 0, stream>>>(qkvb, obuf);

    // 6. xa = x + obuf @ Wproj^T + b
    {
        dim3 g(TOKENS / 64, EMBED / 256);
        gemm_bf16_kernel<<<g, 128, 0, stream>>>(obuf, wproj, proj_b, x,
                                                xa, TOKENS, EMBED, EMBED, 1);
    }

    // 7. LN2 -> hbuf (bf16)
    ln_kernel<<<TOKENS, 256, 0, stream>>>(xa, nullptr, n2_g, n2_b, hbuf);

    // 8. gbuf = gelu(hbuf @ Wfc1^T + b) -> bf16
    {
        dim3 g(TOKENS / 64, MLP_HID / 256);
        gemm_bf16_kernel<<<g, 128, 0, stream>>>(hbuf, wfc1, fc1_b, nullptr,
                                                gbuf, TOKENS, MLP_HID, EMBED, 2);
    }

    // 9. out = xa + gbuf @ Wfc2^T + b
    {
        dim3 g(TOKENS / 64, EMBED / 256);
        gemm_bf16_kernel<<<g, 128, 0, stream>>>(gbuf, wfc2, fc2_b, xa,
                                                out, TOKENS, EMBED, MLP_HID, 1);
    }
}
